// GuideTripletLoss_54511724921507
// MI455X (gfx1250) — matrix operand order repaired and checked by reading, not compiled, not run on hardware
//
#include <hip/hip_runtime.h>
#include <hip/hip_bf16.h>

// ---------------------------------------------------------------------------
// GuideTripletLoss on gfx1250 (MI455X).
// pred: (4,64,192,640) f32, target: (4,1,192,640) i32 -> out: (4,1,192,640) f32
// Single pass over pred (memory-bound: ~130 MB @ 23.3 TB/s ~ 5.6 us floor).
// 25 per-pixel 64-ch dot products computed with v_wmma_f32_16x16x32_f16 from
// an LDS-staged f16 feature halo; normalization deferred to scalar inv-norms.
// ---------------------------------------------------------------------------

typedef __attribute__((ext_vector_type(16))) _Float16 v16h;
typedef __attribute__((ext_vector_type(8)))  _Float16 v8h;
typedef __attribute__((ext_vector_type(8)))  float    v8f;

#define BN 4
#define CN 64
#define HN 192
#define WN 640
#define HWPLANE (HN * WN)          // 122880, channel stride in pred

#define TILE_W 16
#define TILE_H 8
#define HALO 2
#define HLW (TILE_W + 2 * HALO)    // 20 halo cols
#define HLH (TILE_H + 2 * HALO)    // 12 halo rows
#define NPIX (HLW * HLH)           // 240 halo pixels
#define FSTR_H 72                  // feature row stride in halfs (144 B, 16B aligned, bank-spread)
#define FSTR_DW 36                 // same in dwords

static __device__ inline v8h ld8(const _Float16* p) {
    return *(const v8h*)p;         // 16B-aligned -> ds_load_b128
}
static __device__ inline v16h cat16(v8h lo, v8h hi) {
    return __builtin_shufflevector(lo, hi, 0,1,2,3,4,5,6,7,8,9,10,11,12,13,14,15);
}
static __device__ inline v8f wmma_f16(v16h a, v16h b, v8f c) {
    // (neg_a, A, neg_b, B, c_mod, C, reuse_a, reuse_b)
    return __builtin_amdgcn_wmma_f32_16x16x32_f16(false, a, false, b, (short)0, c, false, false);
}

__global__ __launch_bounds__(256)
void GuideTripletLoss_kernel(const float* __restrict__ pred,
                             const int*   __restrict__ target,
                             float*       __restrict__ out) {
    // --- LDS ----------------------------------------------------------------
    __shared__ __align__(16) _Float16 sFeat[NPIX * FSTR_H]; // raw f16 features, padded rows
    __shared__ float sInv[NPIX];                            // per-pixel 1/||x||
    __shared__ int   sLbl[NPIX];                            // labels, -1 = OOB
    __shared__ float sStage[8][2][256];                     // per-wave D-tile staging

    const int t  = threadIdx.x;
    const int bz = blockIdx.z;
    const int r0 = blockIdx.y * TILE_H;
    const int c0 = blockIdx.x * TILE_W;

    // --- Stage 1: load halo, compute sumsq, store raw f16 features ----------
    if (t < NPIX) {
        const int hy = t / HLW, hx = t % HLW;
        const int y = r0 - HALO + hy;
        const int x = c0 - HALO + hx;
        const bool inb = (y >= 0) && (y < HN) && (x >= 0) && (x < WN);
        unsigned* fdw = (unsigned*)sFeat + t * FSTR_DW;
        float ss = 0.0f;
        if (inb) {
            const float* src = pred + (size_t)bz * CN * HWPLANE + (size_t)y * WN + x;
#pragma unroll
            for (int ch = 0; ch < CN; ch += 2) {
                float v0 = src[(size_t)ch * HWPLANE];
                float v1 = src[(size_t)(ch + 1) * HWPLANE];
                ss += v0 * v0 + v1 * v1;
                union { _Float16 h[2]; unsigned u; } pk;
                pk.h[0] = (_Float16)v0;
                pk.h[1] = (_Float16)v1;
                fdw[ch >> 1] = pk.u;
            }
            sLbl[t] = target[((size_t)bz * HN + y) * WN + x];
        } else {
#pragma unroll
            for (int ch = 0; ch < CN; ch += 2) fdw[ch >> 1] = 0u;
            sLbl[t] = -1;
        }
        sInv[t] = rsqrtf(fmaxf(ss, 1e-24f)); // OOB: feat==0 -> sim==0 regardless
    }
    __syncthreads();

    // --- Stage 2: WMMA Gram bands -------------------------------------------
    const int w    = t >> 5;        // wave id = output row within tile
    const int lane = t & 31;
    const int half = lane >> 4;     // lane group (K interleave select)
    const int m    = lane & 15;     // matrix row / col index owned by lane

    const _Float16* fp = sFeat;

    // A fragment: 16 center pixels (row r0+w, cols c0..c0+15), K = 64 in 2 steps.
    // 16-bit A layout (wave32): lanes 0-15 hold K kb+{0..7,16..23}, lanes 16-31 kb+{8..15,24..31}.
    const int pA = (w + HALO) * HLW + (m + HALO);
    const _Float16* ap = fp + pA * FSTR_H;
    const v16h a0 = cat16(ld8(ap + 0  + half * 8), ld8(ap + 16 + half * 8));
    const v16h a1 = cat16(ld8(ap + 32 + half * 8), ld8(ap + 48 + half * 8));

    const int pc    = (w + HALO) * HLW + (m + HALO);
    const float inv_c = sInv[pc];
    const int   lbl_c = sLbl[pc];

    float pos_sum = 0.0f, neg_sum = 0.0f;
    int   pos_cnt = 0;

    const int hx0 = m;                       // B tile0: halo cols 0..15  (c0-2 .. c0+13)
    const int hx1 = min(16 + m, HLW - 1);    // B tile1: halo cols 16..19 (clamped; extra cols unused)

    for (int dy = 0; dy < 5; ++dy) {
        const int hy = w + dy;               // neighbor halo row

        // B layout (wave32): lanes 0-15 = col m, K kb+{0..7,16..23}; lanes 16-31 = col m, K kb+{8..15,24..31}.
        const _Float16* bp0 = fp + (hy * HLW + hx0) * FSTR_H + half * 8;
        const _Float16* bp1 = fp + (hy * HLW + hx1) * FSTR_H + half * 8;

        v8f acc0 = {}, acc1 = {};
        {   // K step 0 (ch 0..31)
            v16h b0 = cat16(ld8(bp0 + 0), ld8(bp0 + 16));
            acc0 = wmma_f16(a0, b0, acc0);
            v16h b1 = cat16(ld8(bp1 + 0), ld8(bp1 + 16));
            acc1 = wmma_f16(a0, b1, acc1);
        }
        {   // K step 1 (ch 32..63)
            v16h b0 = cat16(ld8(bp0 + 32), ld8(bp0 + 48));
            acc0 = wmma_f16(a1, b0, acc0);
            v16h b1 = cat16(ld8(bp1 + 32), ld8(bp1 + 48));
            acc1 = wmma_f16(a1, b1, acc1);
        }

        // Stage D tiles: element (M,N) lives in lane N (+16 if M>=8), vgpr M&7.
#pragma unroll
        for (int v = 0; v < 8; ++v) {
            const int M = v + half * 8;
            sStage[w][0][M * 16 + m] = acc0[v];
            sStage[w][1][M * 16 + m] = acc1[v];
        }
        __syncthreads();

        if (lane < 16) {
            // Lane m = center pixel i; its 5 neighbors this row sit at halo cols m..m+4.
#pragma unroll
            for (int o = 0; o < 5; ++o) {
                const int j = m + o;                       // neighbor halo col (0..19)
                const float d  = sStage[w][j >> 4][m * 16 + (j & 15)];
                const float iv = sInv[hy * HLW + j];
                const int   ln = sLbl[hy * HLW + j];
                const float sim = d * inv_c * iv;
                const float aff = sqrtf(fmaxf(1e-9f, 2.0f - 2.0f * sim));
                if (ln == lbl_c) { pos_sum += aff; pos_cnt += 1; }
                else             { neg_sum += aff; }
            }
        }
        __syncthreads();
    }

    // --- Stage 3: triplet + boundary gate -----------------------------------
    if (lane < 16) {
        const int neg_cnt = 25 - pos_cnt;
        const bool boundary = (pos_cnt >= 4) && (neg_cnt >= 4);
        const float pos_dist = pos_sum / (float)pos_cnt;               // pos_cnt >= 1 (center)
        const float neg_dist = neg_sum / fmaxf((float)neg_cnt, 1e-12f);
        const float trip = fmaxf(0.0f, pos_dist - neg_dist + 0.3f);
        out[((size_t)bz * HN + (r0 + w)) * WN + (c0 + m)] = boundary ? trip : 0.0f;
    }
}

extern "C" void kernel_launch(void* const* d_in, const int* in_sizes, int n_in,
                              void* d_out, int out_size, void* d_ws, size_t ws_size,
                              hipStream_t stream) {
    (void)in_sizes; (void)n_in; (void)d_ws; (void)ws_size; (void)out_size;
    const float* pred   = (const float*)d_in[0];
    const int*   target = (const int*)d_in[1];
    float*       out    = (float*)d_out;

    dim3 grid(WN / TILE_W, HN / TILE_H, BN);   // 40 x 24 x 4
    dim3 block(256);                           // 8 waves -> 8 rows x 16 cols per WG
    GuideTripletLoss_kernel<<<grid, block, 0, stream>>>(pred, target, out);
}
